// TransGNNComplete_64295660421744
// MI455X (gfx1250) — compile-verified, matrix operand order checked by reading, not run on hardware
//
#include <hip/hip_runtime.h>
#include <hip/hip_bf16.h>

#define USER_N 40000
#define ITEM_N 40000
#define NN (USER_N + ITEM_N)
#define DD 128
#define HH 4
#define KK 16
#define EE 1600000
#define LTOK 17          // K+1 tokens
#define XS 136           // LDS row stride (halves) for X tiles
#define QS 132           // Qs row stride (floats)
#define OS 136           // o row stride (halves)

typedef __attribute__((ext_vector_type(16))) __bf16 bf16x16;
typedef __attribute__((ext_vector_type(8)))  float  f32x8;

union Frag { bf16x16 bf; uint4 q[2]; unsigned short u[16]; };

__device__ __forceinline__ unsigned short f2bf(float f) {
  unsigned int u = __float_as_uint(f);
  unsigned int r = (u + 0x7FFFu + ((u >> 16) & 1u)) >> 16;  // round-to-nearest-even
  return (unsigned short)r;
}

// ---------------------------------------------------------------------------
// e = concat(user_emb, item_emb); total = e
// ---------------------------------------------------------------------------
__global__ void init_kernel(const float* __restrict__ ue, const float* __restrict__ ie,
                            float* __restrict__ e, float* __restrict__ total) {
  size_t idx = (size_t)blockIdx.x * blockDim.x + threadIdx.x;
  if (idx >= (size_t)NN * DD) return;
  float v = (idx < (size_t)USER_N * DD) ? ue[idx] : ie[idx - (size_t)USER_N * DD];
  e[idx] = v;
  total[idx] = v;
}

// ---------------------------------------------------------------------------
// Pack 8 weight matrices (u:wq,wk,wv,wo then i:wq,wk,wv,wo) into bf16
// WMMA-B-fragment layout. Fragment (kt,nt): 32 lanes x 16 halves contiguous.
// Per 16-bit B-matrix (32x16) layout: lane L holds column N = L&15; lanes<16
// hold K = {0..7,16..23}, lanes>=16 hold K = {8..15,24..31} of the K-tile.
// ---------------------------------------------------------------------------
__global__ void pack_weights(const float* __restrict__ u_in_w, const float* __restrict__ u_out_w,
                             const float* __restrict__ i_in_w, const float* __restrict__ i_out_w,
                             unsigned short* __restrict__ pw) {
  int idx = blockIdx.x * blockDim.x + threadIdx.x;
  if (idx >= 8 * 16384) return;
  int m   = idx >> 14;
  int fid = idx & 16383;
  const float* src;
  switch (m) {
    case 0: src = u_in_w;                 break;   // wq (user)
    case 1: src = u_in_w + 128 * 128;     break;   // wk
    case 2: src = u_in_w + 2 * 128 * 128; break;   // wv
    case 3: src = u_out_w;                break;   // wo
    case 4: src = i_in_w;                 break;   // wq (item)
    case 5: src = i_in_w + 128 * 128;     break;
    case 6: src = i_in_w + 2 * 128 * 128; break;
    default: src = i_out_w;               break;
  }
  int frag   = fid >> 9;        // 0..31 = kt*8+nt
  int kt     = frag >> 3;
  int nt     = frag & 7;
  int within = fid & 511;
  int lane   = within >> 4;
  int j      = within & 15;
  int hi     = lane >> 4;
  int row16  = lane & 15;
  int sub    = j >> 3;
  int jj     = j & 7;
  int k = kt * 32 + sub * 16 + hi * 8 + jj;
  int n = nt * 16 + row16;
  pw[idx] = f2bf(src[n * 128 + k]);     // B[k][n] = W[n][k]  (X @ W^T)
}

// ---------------------------------------------------------------------------
// tmp[r] += vals * e[c] : one wave per edge, float4 per lane, f32 atomics (L2)
// ---------------------------------------------------------------------------
__global__ void spmm_kernel(const int* __restrict__ rows, const int* __restrict__ cols,
                            const float* __restrict__ vals, const float* __restrict__ e,
                            float* __restrict__ tmp) {
  int gid  = blockIdx.x * blockDim.x + threadIdx.x;
  int eidx = gid >> 5;
  int lane = gid & 31;
  if (eidx >= EE) return;
  int r = rows[eidx];
  int c = cols[eidx];
  float v = vals[eidx];
  const float4* src = (const float4*)(e + (size_t)c * DD);
  float4 x = src[lane];
  float* dst = tmp + (size_t)r * DD + lane * 4;
  atomicAdd(dst + 0, v * x.x);
  atomicAdd(dst + 1, v * x.y);
  atomicAdd(dst + 2, v * x.z);
  atomicAdd(dst + 3, v * x.w);
}

// ---------------------------------------------------------------------------
// Fused MHA block for 16 nodes: stage enriched X (17 tokens x 16 nodes) in LDS
// as bf16, then Q / streamed-K(scores) / softmax / streamed-V(context) / out
// projection, all via v_wmma_f32_16x16x32_bf16. Wave w owns channels
// [16w,16w+16); head h = w/2. Weight B-fragments are hoisted into registers
// before the token loops so the hot loop is ds_load + wmma only.
// ---------------------------------------------------------------------------
__device__ __forceinline__ void load_bfrags(const unsigned short* __restrict__ mat,
                                            int wvid, int lane, Frag bf4[4]) {
  #pragma unroll
  for (int kt = 0; kt < 4; ++kt) {
    const uint4* bp = (const uint4*)&mat[((kt << 3) + wvid) * 512 + lane * 16];
    bf4[kt].q[0] = bp[0];
    bf4[kt].q[1] = bp[1];
  }
}

__launch_bounds__(256, 1)
__global__ void mha_kernel(const float* __restrict__ tmp,
                           const int* __restrict__ samples,
                           const float* __restrict__ pos_emb,
                           const unsigned short* __restrict__ pw,   // 4 packed matrices
                           const float* __restrict__ in_b,          // 3*128 (bq,bk,bv)
                           const float* __restrict__ out_b,         // 128
                           float* __restrict__ e_out,
                           float* __restrict__ total,
                           int node_base) {
  __shared__ __align__(16) unsigned short Xl[LTOK * 16 * XS];   // 73,984 B
  __shared__ float Qs[16 * QS];                                 //  8,448 B
  __shared__ float scr[LTOK * 16 * 8];                          //  8,704 B
  __shared__ float attn[LTOK * 16 * HH];                        //  4,352 B
  __shared__ __align__(16) unsigned short ol[16 * OS];          //  4,352 B

  const int tid    = threadIdx.x;
  const int wvid   = tid >> 5;        // wave 0..7
  const int lane   = tid & 31;
  const int hi     = lane >> 4;       // half-wave
  const int l15    = lane & 15;
  const int base16 = blockIdx.x * 16;

  // ---- stage enriched X (gather tmp rows + pos_emb, fp32 -> bf16) ----
  for (int idx = tid; idx < LTOK * 16 * DD; idx += 256) {
    int r = idx >> 7;          // 0..271 = token*16 + node
    int d = idx & 127;
    int l = r >> 4;
    int i = r & 15;
    int g = node_base + base16 + i;
    int src = (l == 0) ? g : samples[(size_t)g * KK + (l - 1)];
    float v = tmp[(size_t)src * DD + d] + pos_emb[l * DD + d];
    Xl[r * XS + d] = f2bf(v);
  }
  __syncthreads();

  const unsigned short* wqf = pw;
  const unsigned short* wkf = pw + 16384;
  const unsigned short* wvf = pw + 2 * 16384;
  const unsigned short* wof = pw + 3 * 16384;
  const int ch = (wvid << 4) + l15;               // this lane's output channel
  const float rs = 0.17677669529663687f;          // 1/sqrt(dh)

  // ---- Q projection (token 0 tile), pre-scaled, stored fp32 ----
  {
    Frag bw[4];
    load_bfrags(wqf, wvid, lane, bw);
    f32x8 acc = {};
    #pragma unroll
    for (int kt = 0; kt < 4; ++kt) {
      Frag a;
      int k0 = kt * 32 + hi * 8;
      a.q[0] = *(const uint4*)&Xl[l15 * XS + k0];
      a.q[1] = *(const uint4*)&Xl[l15 * XS + k0 + 16];
      acc = __builtin_amdgcn_wmma_f32_16x16x32_bf16(false, a.bf, false, bw[kt].bf,
                                                    (short)0, acc, false, false);
    }
    float bq = in_b[ch];
    #pragma unroll
    for (int r = 0; r < 8; ++r) {
      int i = r + (hi << 3);
      Qs[i * QS + ch] = (acc[r] + bq) * rs;
    }
  }
  __syncthreads();

  // ---- K pass: streamed per token; partial scores via 16-lane reduction ----
  {
    Frag bw[4];
    load_bfrags(wkf, wvid, lane, bw);        // weights live in registers
    float bk = in_b[DD + ch];
    for (int l = 0; l < LTOK; ++l) {
      f32x8 acc = {};
      int m0 = l << 4;
      #pragma unroll
      for (int kt = 0; kt < 4; ++kt) {
        Frag a;
        int k0 = kt * 32 + hi * 8;
        a.q[0] = *(const uint4*)&Xl[(m0 + l15) * XS + k0];
        a.q[1] = *(const uint4*)&Xl[(m0 + l15) * XS + k0 + 16];
        acc = __builtin_amdgcn_wmma_f32_16x16x32_bf16(false, a.bf, false, bw[kt].bf,
                                                      (short)0, acc, false, false);
      }
      #pragma unroll
      for (int r = 0; r < 8; ++r) {
        int i = r + (hi << 3);
        float p = Qs[i * QS + ch] * (acc[r] + bk);
        p += __shfl_xor(p, 1, 16);
        p += __shfl_xor(p, 2, 16);
        p += __shfl_xor(p, 4, 16);
        p += __shfl_xor(p, 8, 16);
        if (l15 == 0) scr[(m0 + i) * 8 + wvid] = p;   // per 16-channel chunk
      }
    }
  }
  __syncthreads();

  // ---- softmax over 17 tokens per (node, head) ----
  if (tid < 64) {
    int i = tid & 15;
    int h = tid >> 4;
    float sc[LTOK];
    float mx = -1e30f;
    for (int l = 0; l < LTOK; ++l) {
      float v = scr[(l * 16 + i) * 8 + 2 * h] + scr[(l * 16 + i) * 8 + 2 * h + 1];
      sc[l] = v;
      mx = fmaxf(mx, v);
    }
    float s = 0.f;
    for (int l = 0; l < LTOK; ++l) { float ex = __expf(sc[l] - mx); sc[l] = ex; s += ex; }
    float inv = 1.f / s;
    for (int l = 0; l < LTOK; ++l) attn[(l * 16 + i) * HH + h] = sc[l] * inv;
  }
  __syncthreads();

  // ---- V pass: streamed per token, scaled by attention (bias folded: sum(a)=1) ----
  {
    Frag bw[4];
    load_bfrags(wvf, wvid, lane, bw);        // weights live in registers
    int h = wvid >> 1;
    f32x8 ctx = {};
    for (int l = 0; l < LTOK; ++l) {
      f32x8 acc = {};
      int m0 = l << 4;
      #pragma unroll
      for (int kt = 0; kt < 4; ++kt) {
        Frag a;
        int k0 = kt * 32 + hi * 8;
        a.q[0] = *(const uint4*)&Xl[(m0 + l15) * XS + k0];
        a.q[1] = *(const uint4*)&Xl[(m0 + l15) * XS + k0 + 16];
        acc = __builtin_amdgcn_wmma_f32_16x16x32_bf16(false, a.bf, false, bw[kt].bf,
                                                      (short)0, acc, false, false);
      }
      #pragma unroll
      for (int r = 0; r < 8; ++r) {
        int i = r + (hi << 3);
        float a_ = attn[(m0 + i) * HH + h];
        ctx[r] += a_ * acc[r];
      }
    }
    float bv = in_b[2 * DD + ch];
    #pragma unroll
    for (int r = 0; r < 8; ++r) {
      int i = r + (hi << 3);
      ol[i * OS + ch] = f2bf(ctx[r] + bv);
    }
  }
  __syncthreads();

  // ---- output projection + bias + residual; write e and accumulate total ----
  {
    Frag bw[4];
    load_bfrags(wof, wvid, lane, bw);
    f32x8 acc = {};
    #pragma unroll
    for (int kt = 0; kt < 4; ++kt) {
      Frag a;
      int k0 = kt * 32 + hi * 8;
      a.q[0] = *(const uint4*)&ol[l15 * OS + k0];
      a.q[1] = *(const uint4*)&ol[l15 * OS + k0 + 16];
      acc = __builtin_amdgcn_wmma_f32_16x16x32_bf16(false, a.bf, false, bw[kt].bf,
                                                    (short)0, acc, false, false);
    }
    float ob = out_b[ch];
    #pragma unroll
    for (int r = 0; r < 8; ++r) {
      int i = r + (hi << 3);
      size_t g = (size_t)(node_base + base16 + i) * DD + ch;
      float v = acc[r] + ob + tmp[g];     // residual += tmp
      e_out[g] = v;
      total[g] += v;
    }
  }
}

// ---------------------------------------------------------------------------
// d_out = [total, total[:USER], total[USER:]] == [total, total]
// ---------------------------------------------------------------------------
__global__ void final_kernel(const float* __restrict__ total, float* __restrict__ out) {
  size_t idx = (size_t)blockIdx.x * blockDim.x + threadIdx.x;
  if (idx >= (size_t)NN * DD) return;
  float v = total[idx];
  out[idx] = v;
  out[(size_t)NN * DD + idx] = v;
}

extern "C" void kernel_launch(void* const* d_in, const int* in_sizes, int n_in,
                              void* d_out, int out_size, void* d_ws, size_t ws_size,
                              hipStream_t stream) {
  (void)in_sizes; (void)n_in; (void)out_size; (void)ws_size;
  const int*   adj_rows = (const int*)d_in[0];
  const int*   adj_cols = (const int*)d_in[1];
  const float* adj_vals = (const float*)d_in[2];
  const int*   samples  = (const int*)d_in[3];
  const float* user_emb = (const float*)d_in[4];
  const float* item_emb = (const float*)d_in[5];
  const float* pos_emb  = (const float*)d_in[6];
  const float* u_in_w   = (const float*)d_in[7];
  const float* u_in_b   = (const float*)d_in[8];
  const float* u_out_w  = (const float*)d_in[9];
  const float* u_out_b  = (const float*)d_in[10];
  const float* i_in_w   = (const float*)d_in[11];
  const float* i_in_b   = (const float*)d_in[12];
  const float* i_out_w  = (const float*)d_in[13];
  const float* i_out_b  = (const float*)d_in[14];

  char* ws = (char*)d_ws;
  const size_t nd = (size_t)NN * DD;
  float* e_buf = (float*)ws;                         // 40.96 MB
  float* tmp   = (float*)(ws + nd * 4);              // 40.96 MB
  float* total = (float*)(ws + 2 * nd * 4);          // 40.96 MB
  unsigned short* pw = (unsigned short*)(ws + 3 * nd * 4);  // 256 KB packed weights

  init_kernel<<<(int)((nd + 255) / 256), 256, 0, stream>>>(user_emb, item_emb, e_buf, total);
  pack_weights<<<(8 * 16384) / 256, 256, 0, stream>>>(u_in_w, u_out_w, i_in_w, i_out_w, pw);

  for (int blk = 0; blk < 2; ++blk) {
    hipMemsetAsync(tmp, 0, nd * 4, stream);
    spmm_kernel<<<(EE * 32) / 256, 256, 0, stream>>>(adj_rows, adj_cols, adj_vals, e_buf, tmp);
    mha_kernel<<<USER_N / 16, 256, 0, stream>>>(tmp, samples, pos_emb, pw,
                                                u_in_b, u_out_b, e_buf, total, 0);
    mha_kernel<<<ITEM_N / 16, 256, 0, stream>>>(tmp, samples, pos_emb, pw + 4 * 16384,
                                                i_in_b, i_out_b, e_buf, total, USER_N);
  }
  final_kernel<<<(int)((nd + 255) / 256), 256, 0, stream>>>(total, (float*)d_out);
}
